// Encoder_z0_ODE_RNN_19069654794517
// MI455X (gfx1250) — compile-verified, matrix operand order checked by reading, not run on hardware
//
#include <hip/hip_runtime.h>
#include <hip/hip_bf16.h>
#include <math.h>

typedef __attribute__((ext_vector_type(16))) _Float16 v16h;
typedef __attribute__((ext_vector_type(8)))  _Float16 v8h;
typedef __attribute__((ext_vector_type(8)))  float    v8f;

#define N_SUBJ   2048
#define N_TPN    128
#define IN_DIM   64
#define HALF_DIM 32
#define LATENT   128
#define N_UNIT   256
#define GIN      320

#define SUBJ_PER_WG 32
#define ROW_TILES   2
#define WAVES       8
#define THREADS     256

// -----------------------------------------------------------------------------
// Branch-free activations on hardware transcendentals (v_exp_f32 / v_log_f32 /
// v_rcp_f32). IEEE inf arithmetic gives correct saturation, no exec-mask churn.
// -----------------------------------------------------------------------------
__device__ __forceinline__ float fast_tanh(float x) {
  float e = __expf(2.0f * x);                       // v_exp_f32
  return 1.0f - 2.0f * __builtin_amdgcn_rcpf(e + 1.0f);
}
__device__ __forceinline__ float fast_sigmoid(float x) {
  return __builtin_amdgcn_rcpf(1.0f + __expf(-x));
}
__device__ __forceinline__ float fast_softplus(float x) {
  // stable: max(x,0) + log(1 + exp(-|x|))
  return fmaxf(x, 0.0f) + __logf(1.0f + __expf(-fabsf(x)));
}

// -----------------------------------------------------------------------------
// Weight prepass: f32 row-major [K][N] -> f16 WMMA-B fragments, lane-major.
// Per ISA 16-bit B layout: lanes 0-15 -> col = lane, K pairs {0..7,16..23};
// lanes 16-31 -> col = lane-16, K pairs {8..15,24..31}. 16 halfs per lane,
// stored contiguously (32 B) so the main kernel does one aligned v16h load.
// Order: ug_w1,rg_w1,ns_w1,ode_w1,ug_w2,rg_w2,ns_w2,ode_w2,tz_w1,tz_w2
// Frags: 160,  160,  160,  64,    64,   64,   128,  64,    128,  128  = 1120
// -----------------------------------------------------------------------------
__global__ void prep_weights(const float* __restrict__ ug_w1, const float* __restrict__ rg_w1,
                             const float* __restrict__ ns_w1, const float* __restrict__ ode_w1,
                             const float* __restrict__ ug_w2, const float* __restrict__ rg_w2,
                             const float* __restrict__ ns_w2, const float* __restrict__ ode_w2,
                             const float* __restrict__ tz_w1, const float* __restrict__ tz_w2,
                             _Float16* __restrict__ ws) {
  const int Ks[10] = {320, 320, 320, 128, 256, 256, 256, 256, 256, 256};
  const int Ns[10] = {256, 256, 256, 256, 128, 128, 256, 128, 256, 256};
  const float* W[10] = {ug_w1, rg_w1, ns_w1, ode_w1, ug_w2, rg_w2, ns_w2, ode_w2, tz_w1, tz_w2};

  int gid = blockIdx.x * blockDim.x + threadIdx.x;
  int fragId = gid >> 5;
  int lane = gid & 31;
  if (fragId >= 1120) return;

  int w = 0, start = 0;
  for (int i = 0; i < 10; ++i) {
    int cnt = (Ks[i] / 32) * (Ns[i] / 16);
    if (fragId < start + cnt) { w = i; break; }
    start += cnt;
  }
  int f   = fragId - start;
  int nct = Ns[w] / 16;
  int kc  = f / nct;
  int nc  = f % nct;
  const float* src = W[w];
  _Float16* dst = ws + (size_t)fragId * 512 + lane * 16;
  int col = nc * 16 + (lane & 15);
  int kb  = kc * 32 + ((lane >= 16) ? 8 : 0);
  #pragma unroll
  for (int v = 0; v < 8; ++v)
    #pragma unroll
    for (int h = 0; h < 2; ++h) {
      int k = kb + ((v < 4) ? (2 * v + h) : (16 + 2 * (v - 4) + h));
      dst[v * 2 + h] = (_Float16)src[(size_t)k * Ns[w] + col];
    }
}

// -----------------------------------------------------------------------------
// One 16x16 output tile, A from LDS (row-major f16), B from pre-swizzled global.
// -----------------------------------------------------------------------------
__device__ __forceinline__ v8f gemm_tile(const _Float16* __restrict__ A, int astride, int rbase,
                                         const _Float16* __restrict__ B, int nct, int nc,
                                         int kchunks, int lane) {
  v8f acc = {};
  int row  = rbase + (lane & 15);
  int ksel = (lane >= 16) ? 8 : 0;
  for (int kc = 0; kc < kchunks; ++kc) {
    const _Float16* ap = A + row * astride + kc * 32 + ksel;
    v8h alo = *(const v8h*)(ap);
    v8h ahi = *(const v8h*)(ap + 16);
    v16h a = __builtin_shufflevector(alo, ahi, 0, 1, 2, 3, 4, 5, 6, 7,
                                     8, 9, 10, 11, 12, 13, 14, 15);
    v16h b = *(const v16h*)(B + ((size_t)(kc * nct + nc)) * 512 + lane * 16);
    acc = __builtin_amdgcn_wmma_f32_16x16x32_f16(false, a, false, b,
                                                 (short)0, acc, false, false);
  }
  return acc;
}

// -----------------------------------------------------------------------------
// Persistent ODE-RNN encoder. One WG = 32 subjects, loops 128 reversed steps.
// -----------------------------------------------------------------------------
__global__ __launch_bounds__(THREADS)
void encoder_kernel(const float* __restrict__ data, const float* __restrict__ obs,
                    const float* __restrict__ ug_b1, const float* __restrict__ ug_b2,
                    const float* __restrict__ rg_b1, const float* __restrict__ rg_b2,
                    const float* __restrict__ ns_b1, const float* __restrict__ ns_b2,
                    const float* __restrict__ ode_b1, const float* __restrict__ ode_b2,
                    const float* __restrict__ tz_b1, const float* __restrict__ tz_b2,
                    const _Float16* __restrict__ wf, float* __restrict__ out) {
  __shared__ _Float16 yc[SUBJ_PER_WG][GIN];        // [mean | std | x] f16 (A source)
  __shared__ _Float16 hbufU[SUBJ_PER_WG][N_UNIT];  // hidden f16 (ode/ug/ns)
  __shared__ _Float16 hbufR[SUBJ_PER_WG][N_UNIT];  // hidden f16 (rg)
  __shared__ float Ymean[SUBJ_PER_WG][LATENT];
  __shared__ float Ystd[SUBJ_PER_WG][LATENT];
  __shared__ float Yode[SUBJ_PER_WG][LATENT];      // mean after Euler step
  __shared__ float Ug[SUBJ_PER_WG][LATENT];
  __shared__ float Rg[SUBJ_PER_WG][LATENT];
  __shared__ float Mk[SUBJ_PER_WG];

  const int tid   = threadIdx.x;
  const int lane  = tid & 31;
  const int wave  = tid >> 5;
  const int sbase = blockIdx.x * SUBJ_PER_WG;

  const _Float16* F_ug1  = wf + (size_t)0   * 512;
  const _Float16* F_rg1  = wf + (size_t)160 * 512;
  const _Float16* F_ns1  = wf + (size_t)320 * 512;
  const _Float16* F_ode1 = wf + (size_t)480 * 512;
  const _Float16* F_ug2  = wf + (size_t)544 * 512;
  const _Float16* F_rg2  = wf + (size_t)608 * 512;
  const _Float16* F_ns2  = wf + (size_t)672 * 512;
  const _Float16* F_ode2 = wf + (size_t)800 * 512;
  const _Float16* F_tz1  = wf + (size_t)864 * 512;
  const _Float16* F_tz2  = wf + (size_t)992 * 512;

  for (int idx = tid; idx < SUBJ_PER_WG * LATENT; idx += THREADS) {
    (&Ymean[0][0])[idx] = 0.0f;
    (&Ystd[0][0])[idx]  = 0.0f;
  }
  __syncthreads();

  for (int i = 0; i < N_TPN; ++i) {
    const int tp = N_TPN - 1 - i;
    const float dt = (i == 0) ? -0.01f : (obs[N_TPN - 1 - i] - obs[N_TPN - i]);

    // ---- phase 0: build yc = [f16(mean) | f16(std) | f16(x)], mask m --------
    for (int idx = tid; idx < SUBJ_PER_WG * LATENT; idx += THREADS) {
      int s = idx >> 7, c = idx & 127;
      yc[s][c]          = (_Float16)Ymean[s][c];
      yc[s][LATENT + c] = (_Float16)Ystd[s][c];
    }
    for (int idx = tid; idx < SUBJ_PER_WG * IN_DIM; idx += THREADS) {
      int s = idx >> 6, c = idx & 63;
      yc[s][2 * LATENT + c] =
          (_Float16)data[(((size_t)(sbase + s)) * N_TPN + tp) * IN_DIM + c];
    }
    if (tid < SUBJ_PER_WG) {
      float sum = 0.0f;
      const float* xp = data + (((size_t)(sbase + tid)) * N_TPN + tp) * IN_DIM + HALF_DIM;
      #pragma unroll 4
      for (int c = 0; c < HALF_DIM; ++c) sum += xp[c];
      Mk[tid] = (sum > 0.0f) ? 1.0f : 0.0f;
      if (tp > 0) {
        const float* nxt = data + (((size_t)(sbase + tid)) * N_TPN + (tp - 1)) * IN_DIM;
        __builtin_prefetch(nxt, 0, 0);
        __builtin_prefetch(nxt + 32, 0, 0);
      }
    }
    __syncthreads();

    // ---- ode hidden: tanh(Ymean @ ode_w1 + b1) -> hbufU ---------------------
    for (int p = wave; p < ROW_TILES * 16; p += WAVES) {
      int rt = p >> 4, nc = p & 15;
      v8f acc = gemm_tile(&yc[0][0], GIN, rt * 16, F_ode1, 16, nc, 4, lane);
      int feat = nc * 16 + (lane & 15);
      float bv = ode_b1[feat];
      int sb = rt * 16 + ((lane >= 16) ? 8 : 0);
      #pragma unroll
      for (int g = 0; g < 8; ++g) hbufU[sb + g][feat] = (_Float16)fast_tanh(acc[g] + bv);
    }
    __syncthreads();

    // ---- Euler step: yode = Ymean + dt*(h @ ode_w2 + b2) --------------------
    for (int p = wave; p < ROW_TILES * 8; p += WAVES) {
      int rt = p >> 3, nc = p & 7;
      v8f acc = gemm_tile(&hbufU[0][0], N_UNIT, rt * 16, F_ode2, 8, nc, 8, lane);
      int feat = nc * 16 + (lane & 15);
      float bv = ode_b2[feat];
      int sb = rt * 16 + ((lane >= 16) ? 8 : 0);
      #pragma unroll
      for (int g = 0; g < 8; ++g) {
        float y = Ymean[sb + g][feat] + dt * (acc[g] + bv);
        Yode[sb + g][feat] = y;
        yc[sb + g][feat]   = (_Float16)y;   // yc mean slot <- ODE-updated mean
      }
    }
    __syncthreads();

    // ---- fused gate layer 1: tanh(yc@{ug_w1|rg_w1}+b) -> hbufU|hbufR --------
    for (int p = wave; p < ROW_TILES * 16 * 2; p += WAVES) {
      int sel = p >> 5;                    // 0 = update gate, 1 = reset gate
      int rt  = (p >> 4) & 1, nc = p & 15;
      const _Float16* Fb = sel ? F_rg1 : F_ug1;
      const float*    Bb = sel ? rg_b1 : ug_b1;
      v8f acc = gemm_tile(&yc[0][0], GIN, rt * 16, Fb, 16, nc, 10, lane);
      int feat = nc * 16 + (lane & 15);
      float bv = Bb[feat];
      int sb = rt * 16 + ((lane >= 16) ? 8 : 0);
      _Float16 (*dst)[N_UNIT] = sel ? hbufR : hbufU;
      #pragma unroll
      for (int g = 0; g < 8; ++g) dst[sb + g][feat] = (_Float16)fast_tanh(acc[g] + bv);
    }
    __syncthreads();

    // ---- fused gate layer 2: sigmoid(h@{ug_w2|rg_w2}+b) -> Ug|Rg ------------
    for (int p = wave; p < ROW_TILES * 8 * 2; p += WAVES) {
      int sel = p >> 4;
      int q   = p & 15;
      int rt  = q >> 3, nc = q & 7;
      const _Float16* Fb = sel ? F_rg2 : F_ug2;
      const float*    Bb = sel ? rg_b2 : ug_b2;
      const _Float16* Ab = sel ? &hbufR[0][0] : &hbufU[0][0];
      v8f acc = gemm_tile(Ab, N_UNIT, rt * 16, Fb, 8, nc, 8, lane);
      int feat = nc * 16 + (lane & 15);
      float bv = Bb[feat];
      int sb = rt * 16 + ((lane >= 16) ? 8 : 0);
      float (*dst)[LATENT] = sel ? Rg : Ug;
      #pragma unroll
      for (int g = 0; g < 8; ++g) dst[sb + g][feat] = fast_sigmoid(acc[g] + bv);
    }
    __syncthreads();

    // ---- yc2 = [mean*r | std*r | x] in place --------------------------------
    for (int idx = tid; idx < SUBJ_PER_WG * 2 * LATENT; idx += THREADS) {
      int s = idx >> 8, c = idx & 255;
      float rv = Rg[s][c & 127];
      yc[s][c] = (_Float16)((float)yc[s][c] * rv);
    }
    __syncthreads();

    // ---- new state layer 1: tanh(yc2@ns_w1+b) -> hbufU ----------------------
    for (int p = wave; p < ROW_TILES * 16; p += WAVES) {
      int rt = p >> 4, nc = p & 15;
      v8f acc = gemm_tile(&yc[0][0], GIN, rt * 16, F_ns1, 16, nc, 10, lane);
      int feat = nc * 16 + (lane & 15);
      float bv = ns_b1[feat];
      int sb = rt * 16 + ((lane >= 16) ? 8 : 0);
      #pragma unroll
      for (int g = 0; g < 8; ++g) hbufU[sb + g][feat] = (_Float16)fast_tanh(acc[g] + bv);
    }
    __syncthreads();

    // ---- new state layer 2 + fused GRU mix + observation mask ---------------
    for (int p = wave; p < ROW_TILES * 16; p += WAVES) {
      int rt = p >> 4, nc = p & 15;
      v8f acc = gemm_tile(&hbufU[0][0], N_UNIT, rt * 16, F_ns2, 16, nc, 8, lane);
      int feat = nc * 16 + (lane & 15);
      float bv = ns_b2[feat];
      int sb = rt * 16 + ((lane >= 16) ? 8 : 0);
      #pragma unroll
      for (int g = 0; g < 8; ++g) {
        int s = sb + g;
        float val = acc[g] + bv;
        float mm  = Mk[s];
        if (feat < LATENT) {
          float uu = Ug[s][feat], yo = Yode[s][feat];
          float mixed = (1.0f - uu) * val + uu * yo;
          Ymean[s][feat] = mm * mixed + (1.0f - mm) * yo;
        } else {
          int f2 = feat - LATENT;
          float sp = fast_softplus(val) + 1e-6f;
          float uu = Ug[s][f2], ys = Ystd[s][f2];
          float mixed = (1.0f - uu) * sp + uu * ys;
          Ystd[s][f2] = fabsf(mm * mixed + (1.0f - mm) * ys);
        }
      }
    }
    __syncthreads();
  }

  // ---- transform_z0: out = tanh([mean|std]@tz_w1+b)@tz_w2+b -----------------
  for (int idx = tid; idx < SUBJ_PER_WG * LATENT; idx += THREADS) {
    int s = idx >> 7, c = idx & 127;
    yc[s][c]          = (_Float16)Ymean[s][c];
    yc[s][LATENT + c] = (_Float16)Ystd[s][c];
  }
  __syncthreads();
  for (int p = wave; p < ROW_TILES * 16; p += WAVES) {
    int rt = p >> 4, nc = p & 15;
    v8f acc = gemm_tile(&yc[0][0], GIN, rt * 16, F_tz1, 16, nc, 8, lane);
    int feat = nc * 16 + (lane & 15);
    float bv = tz_b1[feat];
    int sb = rt * 16 + ((lane >= 16) ? 8 : 0);
    #pragma unroll
    for (int g = 0; g < 8; ++g) hbufU[sb + g][feat] = (_Float16)fast_tanh(acc[g] + bv);
  }
  __syncthreads();
  for (int p = wave; p < ROW_TILES * 16; p += WAVES) {
    int rt = p >> 4, nc = p & 15;
    v8f acc = gemm_tile(&hbufU[0][0], N_UNIT, rt * 16, F_tz2, 16, nc, 8, lane);
    int feat = nc * 16 + (lane & 15);
    float bv = tz_b2[feat];
    int sb = rt * 16 + ((lane >= 16) ? 8 : 0);
    #pragma unroll
    for (int g = 0; g < 8; ++g) {
      int gs = sbase + sb + g;
      float val = acc[g] + bv;
      if (feat < LATENT)
        out[(size_t)gs * LATENT + feat] = val;                                   // mean_z0
      else
        out[(size_t)N_SUBJ * LATENT + (size_t)gs * LATENT + (feat - LATENT)] =
            fabsf(val);                                                          // std_z0
    }
  }
}

extern "C" void kernel_launch(void* const* d_in, const int* in_sizes, int n_in,
                              void* d_out, int out_size, void* d_ws, size_t ws_size,
                              hipStream_t stream) {
  const float* data   = (const float*)d_in[0];
  const float* obs    = (const float*)d_in[1];
  const float* ug_w1  = (const float*)d_in[2];
  const float* ug_b1  = (const float*)d_in[3];
  const float* ug_w2  = (const float*)d_in[4];
  const float* ug_b2  = (const float*)d_in[5];
  const float* rg_w1  = (const float*)d_in[6];
  const float* rg_b1  = (const float*)d_in[7];
  const float* rg_w2  = (const float*)d_in[8];
  const float* rg_b2  = (const float*)d_in[9];
  const float* ns_w1  = (const float*)d_in[10];
  const float* ns_b1  = (const float*)d_in[11];
  const float* ns_w2  = (const float*)d_in[12];
  const float* ns_b2  = (const float*)d_in[13];
  const float* ode_w1 = (const float*)d_in[14];
  const float* ode_b1 = (const float*)d_in[15];
  const float* ode_w2 = (const float*)d_in[16];
  const float* ode_b2 = (const float*)d_in[17];
  const float* tz_w1  = (const float*)d_in[18];
  const float* tz_b1  = (const float*)d_in[19];
  const float* tz_w2  = (const float*)d_in[20];
  const float* tz_b2  = (const float*)d_in[21];

  _Float16* wf = (_Float16*)d_ws;   // 1120 frags * 512 halfs = 1.09 MB
  float* out   = (float*)d_out;

  prep_weights<<<(1120 * 32 + 255) / 256, 256, 0, stream>>>(
      ug_w1, rg_w1, ns_w1, ode_w1, ug_w2, rg_w2, ns_w2, ode_w2, tz_w1, tz_w2, wf);

  encoder_kernel<<<N_SUBJ / SUBJ_PER_WG, THREADS, 0, stream>>>(
      data, obs, ug_b1, ug_b2, rg_b1, rg_b2, ns_b1, ns_b2,
      ode_b1, ode_b2, tz_b1, tz_b2, wf, out);
}
